// NeuralMemory_64364379898522
// MI455X (gfx1250) — matrix operand order repaired and checked by reading, not compiled, not run on hardware
//
#include <hip/hip_runtime.h>
#include <hip/hip_bf16.h>
#include <math.h>

// ---------- types ----------
typedef __attribute__((ext_vector_type(16))) __bf16 v16bf;
typedef __attribute__((ext_vector_type(8)))  float  v8f;
typedef __attribute__((ext_vector_type(2)))  float  v2f;
typedef unsigned int u32x4 __attribute__((ext_vector_type(4)));
typedef int          i32x4v __attribute__((ext_vector_type(4)));
typedef int          i32x8v __attribute__((ext_vector_type(8)));

__device__ __forceinline__ __bf16 f2bf(float x) {
    unsigned u = __builtin_bit_cast(unsigned, x);
    unsigned r = (u + 0x7FFFu + ((u >> 16) & 1u)) >> 16;
    unsigned short s = (unsigned short)r;
    return __builtin_bit_cast(__bf16, s);
}

__device__ __forceinline__ v8f wmma_bf16(v16bf a, v16bf b, v8f c) {
    return __builtin_amdgcn_wmma_f32_16x16x32_bf16(false, a, false, b, (short)0, c, false, false);
}
__device__ __forceinline__ v8f wmma_f32(v2f a, v2f b, v8f c) {
    return __builtin_amdgcn_wmma_f32_16x16x4_f32(false, a, false, b, (short)0, c, false, false);
}

__device__ __forceinline__ v16bf load_frag(const __bf16* p0, const __bf16* p1) {
    union { v16bf v; uint4 u[2]; } t;
    t.u[0] = *reinterpret_cast<const uint4*>(p0);
    t.u[1] = *reinterpret_cast<const uint4*>(p1);
    return t.v;
}

// ---------- constants ----------
#define BB     4
#define SS     2048
#define DM     1024
#define DMEM   256
#define TT     16
#define NCHUNK 128
#define MROWS  (BB*SS)   // 8192
#define SLICE  128       // state-column slice per workgroup (2 slices / batch)

// ---------- Tensor Data Mover (gfx1250) ----------
#if __has_builtin(__builtin_amdgcn_tensor_load_to_lds)
#define HAVE_TDM 1
#else
#define HAVE_TDM 0
#endif
#if __has_include(<hip/amd_detail/amd_gfx1250_TDM.h>)
#define TDM_6ARG 1
#else
#define TDM_6ARG 0
#endif

// Issue one 2-D f32 tile DMA (tile_w x tile_h elements, row stride in elements)
// global -> LDS.  D# packed per CDNA5 ISA 8.3/8.4 (count=1, data_size=4B, type=2).
__device__ __forceinline__ void tdm_load_tile_f32(unsigned lds_byte_addr,
                                                  const float* gptr,
                                                  unsigned tile_w, unsigned tile_h,
                                                  unsigned row_stride) {
#if HAVE_TDM
    unsigned long long ga = (unsigned long long)(size_t)gptr;
    u32x4 g0;
    g0[0] = 1u;                                            // count=1, user mode
    g0[1] = lds_byte_addr;                                 // lds_addr
    g0[2] = (unsigned)ga;                                  // global_addr[31:0]
    g0[3] = (unsigned)((ga >> 32) & 0x01FFFFFFu) | (2u << 30);  // ga[56:32] | type=2
    i32x8v g1;
    g1[0] = (int)(2u << 16);                               // wg_mask=0, data_size=4B
    g1[1] = (int)(tile_w << 16);                           // tensor_dim0[15:0]
    g1[2] = (int)((tile_w >> 16) | (tile_h << 16));        // dim0 hi | tensor_dim1 lo
    g1[3] = (int)((tile_h >> 16) | (tile_w << 16));        // dim1 hi | tile_dim0
    g1[4] = (int)tile_h;                                   // tile_dim1 | tile_dim2=0
    g1[5] = (int)row_stride;                               // tensor_dim0_stride lo32
    g1[6] = 0;
    g1[7] = 0;
    i32x4v z4 = {};
#if TDM_6ARG
    i32x8v z8 = {};
    __builtin_amdgcn_tensor_load_to_lds(g0, g1, z4, z4, z8, 0);
#else
    __builtin_amdgcn_tensor_load_to_lds(g0, g1, z4, z4, 0);
#endif
#endif
}

__device__ __forceinline__ void wait_tensorcnt0() {
#if __has_builtin(__builtin_amdgcn_s_wait_tensorcnt)
    __builtin_amdgcn_s_wait_tensorcnt(0);
#else
    asm volatile("s_wait_tensorcnt 0x0" ::: "memory");
#endif
}
__device__ __forceinline__ void wait_tensorcnt2() {
#if __has_builtin(__builtin_amdgcn_s_wait_tensorcnt)
    __builtin_amdgcn_s_wait_tensorcnt(2);
#else
    asm volatile("s_wait_tensorcnt 0x2" ::: "memory");
#endif
}

// ---------- prep kernels ----------
__global__ __launch_bounds__(256) void cvt_x_kernel(const float* __restrict__ x,
                                                    __bf16* __restrict__ xb) {
    int i = blockIdx.x * 256 + threadIdx.x;   // exact: 8388608
    xb[i] = f2bf(x[i]);
}

__global__ __launch_bounds__(256) void build_wt_kernel(const float* __restrict__ Wk,
                                                       const float* __restrict__ Wv,
                                                       const float* __restrict__ Wq,
                                                       __bf16* __restrict__ WT) {
    int i   = blockIdx.x * 256 + threadIdx.x; // exact: 3*262144
    int mat = i >> 18;
    int rem = i & 262143;
    int n   = rem >> 10;
    int kk  = rem & 1023;
    const float* W = (mat == 0) ? Wk : ((mat == 1) ? Wv : Wq);
    WT[i] = f2bf(W[(size_t)kk * DMEM + n]);
}

__global__ __launch_bounds__(256) void build_wot_kernel(const float* __restrict__ Wo,
                                                        __bf16* __restrict__ WoT) {
    int i = blockIdx.x * 256 + threadIdx.x;   // exact: 262144
    int n = i >> 8;
    int k = i & 255;
    WoT[i] = f2bf(Wo[(size_t)k * DM + n]);
}

__global__ __launch_bounds__(256) void gate_kernel(const float* __restrict__ x,
                                                   const float* __restrict__ Wg,
                                                   const float* __restrict__ bg,
                                                   float* __restrict__ sur) {
    __shared__ float red[256];
    int row = blockIdx.x;
    const float* xr = x + (size_t)row * DM;
    float s = 0.f;
    for (int i = threadIdx.x; i < DM; i += 256) s += xr[i] * Wg[i];
    red[threadIdx.x] = s;
    __syncthreads();
    for (int off = 128; off; off >>= 1) {
        if (threadIdx.x < off) red[threadIdx.x] += red[threadIdx.x + off];
        __syncthreads();
    }
    if (threadIdx.x == 0) {
        float z = red[0] + bg[0];
        sur[row] = 1.0f / (1.0f + __expf(-z));
    }
}

// ---------- projection GEMM (bf16 WMMA) ----------
__global__ __launch_bounds__(256) void proj_gemm_kernel(const __bf16* __restrict__ xb,
                                                        const __bf16* __restrict__ WT,
                                                        const float* __restrict__ bk,
                                                        const float* __restrict__ bv,
                                                        const float* __restrict__ bq,
                                                        float* __restrict__ kvq) {
    int wid  = threadIdx.x >> 5;
    int lane = threadIdx.x & 31;
    int half = lane >> 4, l16 = lane & 15;
    int mat = blockIdx.z;
    int M0  = blockIdx.x * 128 + wid * 16;
    int N0  = blockIdx.y * 16;

    const __bf16* wt   = WT + (size_t)mat * DMEM * DM;
    const float*  bias = (mat == 0) ? bk : ((mat == 1) ? bv : bq);

    v8f c = {};
    const __bf16* abase = xb + (size_t)(M0 + l16) * DM + half * 16;
    const __bf16* bbase = wt + (size_t)(N0 + l16) * DM + half * 16;
    for (int kk = 0; kk < DM; kk += 32) {
        v16bf a = load_frag(abase + kk, abase + kk + 8);
        v16bf b = load_frag(bbase + kk, bbase + kk + 8);
        c = wmma_bf16(a, b, c);
    }
    float bb = bias[N0 + l16];
    float* outp = kvq + (size_t)mat * MROWS * DMEM;
    for (int r = 0; r < 8; ++r) {
        int row = M0 + r + 8 * half;
        outp[(size_t)row * DMEM + N0 + l16] = c[r] + bb;
    }
}

// ---------- chunked recurrent scan (f32 WMMA), 2 column-slices per batch ----------
__global__ __launch_bounds__(256) void scan_kernel(const float* __restrict__ kvq,
                                                   const float* __restrict__ sur,
                                                   const float* __restrict__ state_in,
                                                   const float* __restrict__ decay_p,
                                                   __bf16* __restrict__ mo_bf,
                                                   float* __restrict__ state_out) {
    extern __shared__ float smem[];
    float* S     = smem;                  // 256 x 128 slice          (32768)
    float* QcB   = smem + 32768;          // 2 x (16x256) Q buffers   ( 8192)
    float* KcB   = smem + 40960;          // 2 x (16x128) K buffers   ( 4096)
    float* Vw    = smem + 45056;          // 16x256 weighted V        ( 4096)
    float* Apart = smem + 49152;          // 8 waves x 256 partial A  ( 2048)
    float* Aw    = smem + 51200;          // masked/weighted A        (  256)
    float* dpow  = smem + 51456;          // decay^0..16              (   17)
    float* dinv  = smem + 51473;          // decay^(t-16)             (   16)
    float* sj    = smem + 51489;          // surprise chunk           (   16)

    const int b       = blockIdx.x >> 1;
    const int sl      = blockIdx.x & 1;
    const int colbase = sl * SLICE;
    const int tid  = threadIdx.x;
    const int wid  = tid >> 5;
    const int lane = tid & 31;
    const int half = lane >> 4;
    const int l16  = lane & 15;
    const float decay = decay_p[0];

    if (tid == 0) {
        float p = 1.f;
        for (int i = 0; i <= 16; ++i) { dpow[i] = p; p *= decay; }
        float inv16 = 1.f / dpow[16];
        for (int t = 0; t < 16; ++t) dinv[t] = dpow[t] * inv16;
    }
    for (int i = tid; i < DMEM * SLICE; i += 256) {
        int d = i >> 7, e = i & (SLICE - 1);
        S[i] = state_in[(size_t)b * DMEM * DMEM + (size_t)d * DMEM + colbase + e];
    }

    const float* kb = kvq + (size_t)0 * MROWS * DMEM + (size_t)b * SS * DMEM;
    const float* vb = kvq + (size_t)1 * MROWS * DMEM + (size_t)b * SS * DMEM;
    const float* qb = kvq + (size_t)2 * MROWS * DMEM + (size_t)b * SS * DMEM;

#if HAVE_TDM
    if (wid == 0) {   // preload chunk 0 via Tensor Data Mover
        tdm_load_tile_f32((unsigned)(size_t)QcB, qb, 256, 16, 256);
        tdm_load_tile_f32((unsigned)(size_t)KcB, kb + colbase, 128, 16, 256);
    }
#endif
    __syncthreads();

    for (int c = 0; c < NCHUNK; ++c) {
        const int cur = c & 1;
        float* Qc = QcB + cur * 4096;
        float* Kc = KcB + cur * 2048;

        if (tid < 16) sj[tid] = sur[(size_t)b * SS + c * TT + tid];
        if (c + 1 < NCHUNK)
            __builtin_prefetch(vb + (size_t)((c + 1) * TT) * DMEM + tid, 0, 1);

#if HAVE_TDM
        if (wid == 0) {
            if (c + 1 < NCHUNK) {   // double-buffer: DMA chunk c+1, wait only for chunk c
                int nxt = (c + 1) & 1;
                const float* qn = qb + (size_t)((c + 1) * TT) * DMEM;
                const float* kn = kb + (size_t)((c + 1) * TT) * DMEM + colbase;
                tdm_load_tile_f32((unsigned)(size_t)(QcB + nxt * 4096), qn, 256, 16, 256);
                tdm_load_tile_f32((unsigned)(size_t)(KcB + nxt * 2048), kn, 128, 16, 256);
                wait_tensorcnt2();   // TENSORcnt is in-order: oldest pair (chunk c) done
            } else {
                wait_tensorcnt0();
            }
        }
        __syncthreads();            // Q/K tile + sj ready
        for (int r = 0; r < TT; ++r)
            Vw[r * DMEM + tid] = vb[(size_t)(c * TT + r) * DMEM + tid] * (sj[r] * dpow[15 - r]);
        __syncthreads();            // Vw ready
#else
        __syncthreads();            // sj ready
        for (int r = 0; r < TT; ++r) {
            size_t g = (size_t)(c * TT + r) * DMEM;
            Qc[r * DMEM + tid] = qb[g + tid];
            Vw[r * DMEM + tid] = vb[g + tid] * (sj[r] * dpow[15 - r]);
            if (tid < SLICE) Kc[r * SLICE + tid] = kb[g + colbase + tid];
        }
        __syncthreads();
#endif

        // --- Phase A: A'[t,j] = q_t . (w_j v_j); each wave takes a 32-wide K slice ---
        {
            v8f ca = {};
            int kbase = wid * 32 + 2 * half;
            const float* qrow = Qc + l16 * DMEM;
            const float* vrow = Vw + l16 * DMEM;
            for (int kk = 0; kk < 32; kk += 4) {
                v2f a, bf;
                a.x  = qrow[kbase + kk];  a.y  = qrow[kbase + kk + 1];
                bf.x = vrow[kbase + kk];  bf.y = vrow[kbase + kk + 1];
                ca = wmma_f32(a, bf, ca);
            }
            for (int r = 0; r < 8; ++r)
                Apart[wid * 256 + (r + 8 * half) * 16 + l16] = ca[r];
        }
        __syncthreads();
        {   // reduce + causal mask:  Aw[t,j] = (j<t) ? sum * decay^{t-16} : 0
            int t = tid >> 4, j = tid & 15;
            float s = 0.f;
            for (int w2 = 0; w2 < 8; ++w2) s += Apart[w2 * 256 + tid];
            Aw[tid] = (j < t) ? s * dinv[t] : 0.f;
        }
        __syncthreads();

        // --- Phase B: out = diag(decay^t)(Qc @ S0) + Aw @ Kc ; one N-tile per wave ---
        {
            int N0 = wid * 16;
            v8f cg = {};
            const float* qrow = Qc + l16 * DMEM + 2 * half;
            for (int kk = 0; kk < DMEM; kk += 4) {
                v2f a, bf;
                a.x = qrow[kk]; a.y = qrow[kk + 1];
                int d0 = kk + 2 * half;
                bf.x = S[d0 * SLICE + N0 + l16];
                bf.y = S[(d0 + 1) * SLICE + N0 + l16];
                cg = wmma_f32(a, bf, cg);
            }
            for (int r = 0; r < 8; ++r) cg[r] *= dpow[r + 8 * half];
            const float* arow = Aw + l16 * 16 + 2 * half;
            for (int kk = 0; kk < 16; kk += 4) {
                v2f a, bf;
                a.x = arow[kk]; a.y = arow[kk + 1];
                int d0 = kk + 2 * half;
                bf.x = Kc[d0 * SLICE + N0 + l16];
                bf.y = Kc[(d0 + 1) * SLICE + N0 + l16];
                cg = wmma_f32(a, bf, cg);
            }
            for (int r = 0; r < 8; ++r) {
                int t = r + 8 * half;
                mo_bf[((size_t)b * SS + c * TT + t) * DMEM + colbase + N0 + l16] = f2bf(cg[r]);
            }
        }
        __syncthreads();

        // --- Phase C: S = decay^16 * S + Vw^T @ Kc ; 16 tiles per wave ---
        {
            float d16 = dpow[16];
            for (int ti = wid * 16; ti < wid * 16 + 16; ++ti) {
                int M0 = (ti >> 3) * 16;
                int N0 = (ti & 7) * 16;
                v8f cs;
                for (int r = 0; r < 8; ++r)
                    cs[r] = d16 * S[(M0 + r + 8 * half) * SLICE + N0 + l16];
                for (int kk = 0; kk < 16; kk += 4) {
                    int d0 = kk + 2 * half;
                    v2f a, bf;
                    a.x  = Vw[d0 * DMEM + M0 + l16];
                    a.y  = Vw[(d0 + 1) * DMEM + M0 + l16];
                    bf.x = Kc[d0 * SLICE + N0 + l16];
                    bf.y = Kc[(d0 + 1) * SLICE + N0 + l16];
                    cs = wmma_f32(a, bf, cs);
                }
                for (int r = 0; r < 8; ++r)
                    S[(M0 + r + 8 * half) * SLICE + N0 + l16] = cs[r];
            }
        }
        __syncthreads();
    }

    for (int i = tid; i < DMEM * SLICE; i += 256) {
        int d = i >> 7, e = i & (SLICE - 1);
        state_out[(size_t)b * DMEM * DMEM + (size_t)d * DMEM + colbase + e] = S[i];
    }
}

// ---------- output GEMM (bf16 WMMA) ----------
__global__ __launch_bounds__(256) void out_gemm_kernel(const __bf16* __restrict__ mo,
                                                       const __bf16* __restrict__ WoT,
                                                       const float* __restrict__ bo,
                                                       float* __restrict__ y) {
    int wid  = threadIdx.x >> 5;
    int lane = threadIdx.x & 31;
    int half = lane >> 4, l16 = lane & 15;
    int M0 = blockIdx.x * 128 + wid * 16;
    int N0 = blockIdx.y * 16;
    v8f c = {};
    const __bf16* ab = mo  + (size_t)(M0 + l16) * DMEM + half * 16;
    const __bf16* bb = WoT + (size_t)(N0 + l16) * DMEM + half * 16;
    for (int kk = 0; kk < DMEM; kk += 32) {
        v16bf a = load_frag(ab + kk, ab + kk + 8);
        v16bf b = load_frag(bb + kk, bb + kk + 8);
        c = wmma_bf16(a, b, c);
    }
    float bias = bo[N0 + l16];
    for (int r = 0; r < 8; ++r) {
        int row = M0 + r + 8 * half;
        y[(size_t)row * DM + N0 + l16] = c[r] + bias;
    }
}

// ---------- launcher ----------
extern "C" void kernel_launch(void* const* d_in, const int* in_sizes, int n_in,
                              void* d_out, int out_size, void* d_ws, size_t ws_size,
                              hipStream_t stream) {
    (void)in_sizes; (void)n_in; (void)out_size; (void)ws_size;
    const float* x     = (const float*)d_in[0];
    const float* state = (const float*)d_in[1];
    const float* Wk    = (const float*)d_in[2];
    const float* bk    = (const float*)d_in[3];
    const float* Wv    = (const float*)d_in[4];
    const float* bv    = (const float*)d_in[5];
    const float* Wq    = (const float*)d_in[6];
    const float* bq    = (const float*)d_in[7];
    const float* Wo    = (const float*)d_in[8];
    const float* bo    = (const float*)d_in[9];
    const float* Wg    = (const float*)d_in[10];
    const float* bg    = (const float*)d_in[11];
    const float* decay = (const float*)d_in[12];

    char* ws = (char*)d_ws;
    __bf16* xb  = (__bf16*)(ws + 0);           // 16,777,216
    __bf16* WT  = (__bf16*)(ws + 16777216);    //  1,572,864
    __bf16* WoT = (__bf16*)(ws + 18350080);    //    524,288
    float*  kvq = (float*) (ws + 18874368);    // 25,165,824
    float*  sur = (float*) (ws + 44040192);    //     32,768
    __bf16* mo  = (__bf16*)(ws + 44072960);    //  4,194,304

    float* y         = (float*)d_out;
    float* state_out = (float*)d_out + (size_t)BB * SS * DM;

    cvt_x_kernel    <<<32768, 256, 0, stream>>>(x, xb);
    build_wt_kernel <<<3072,  256, 0, stream>>>(Wk, Wv, Wq, WT);
    build_wot_kernel<<<1024,  256, 0, stream>>>(Wo, WoT);
    gate_kernel     <<<8192,  256, 0, stream>>>(x, Wg, bg, sur);

    proj_gemm_kernel<<<dim3(64, 16, 3), 256, 0, stream>>>(xb, WT, bk, bv, bq, kvq);

    // 8 workgroups: (batch, column-slice); ~206 KB dynamic LDS each
    scan_kernel<<<BB * 2, 256, 51520 * sizeof(float), stream>>>(kvq, sur, state, decay,
                                                                mo, state_out);

    out_gemm_kernel<<<dim3(64, 64), 256, 0, stream>>>(mo, WoT, bo, y);
}
